// Autoencoder_2121713845208
// MI455X (gfx1250) — matrix operand order repaired and checked by reading, not compiled, not run on hardware
//
#include <hip/hip_runtime.h>
#include <hip/hip_bf16.h>

// ---------------------------------------------------------------------------
// CDNA5 (gfx1250) TopK sparse autoencoder forward.
// B=8192, D=2048, L=16384, k=32.
// Two dense bf16 WMMA GEMMs (compute-bound), double-buffered LDS staging,
// LDS-row-cached top-32 select, and an L2-resident sparse gather decode.
// ---------------------------------------------------------------------------

typedef __attribute__((ext_vector_type(16))) __bf16 v16bf;
typedef __attribute__((ext_vector_type(8)))  float  v8f;

union FragU {
    v16bf v;
    uint4 q[2];
};

struct __align__(8) BH4 { __bf16 x, y, z, w; };

#define BM 128
#define BN 128
#define BK 32
#define KPAD 40   // bf16 units; 80-byte row stride (16B aligned, bank-spread)

__device__ __forceinline__ float f4c(const float4& v, int c) {
    return c == 0 ? v.x : c == 1 ? v.y : c == 2 ? v.z : v.w;
}

// C = act( (A [- kbias_over_K]) * Bmat + cbias_over_cols )
// A: [M,K] f32 row-major. Bmat: [K,N] f32 row-major. C: [M,N] f32 row-major.
template <bool SUBK, bool RELU>
__global__ __launch_bounds__(256)
void gemm_bf16_wmma(const float* __restrict__ A,
                    const float* __restrict__ Bmat,
                    float* __restrict__ C,
                    const float* __restrict__ kbias,   // length K (nullable)
                    const float* __restrict__ cbias,   // length N
                    int M, int N, int K) {
    __shared__ __bf16 As[2][BM][KPAD];   // [buf][m][k]
    __shared__ __bf16 Bs[2][BN][KPAD];   // [buf][n][k]  (B tile transposed)

    const int tid   = threadIdx.x;
    const int lane  = tid & 31;
    const int wid   = tid >> 5;       // 8 waves
    const int waveM = wid >> 2;       // 0..1 -> 64-row strip
    const int waveN = wid & 3;        // 0..3 -> 32-col strip
    const int half  = lane >> 4;      // hi/lo 16 lanes
    const int l16   = lane & 15;

    const int blockM = blockIdx.y * BM;
    const int blockN = blockIdx.x * BN;

    v8f acc[4][2];
#pragma unroll
    for (int i = 0; i < 4; ++i)
#pragma unroll
        for (int j = 0; j < 2; ++j)
            acc[i][j] = v8f{0.f, 0.f, 0.f, 0.f, 0.f, 0.f, 0.f, 0.f};

    // staging coordinates (256 threads)
    const int arow = tid >> 3;            // 0..31 (x4 passes -> 128 rows)
    const int akc  = (tid & 7) << 2;      // A k-chunk of 4
    const int bkc  = (tid >> 5) << 2;     // B k-chunk of 4 (0..28)
    const int bnc  = (tid & 31) << 2;     // B n-chunk of 4 (0..124)

    float4 ra[4];   // staged A rows (arow + 32p, k akc..akc+3)
    float4 rb[4];   // staged B rows (k bkc..bkc+3, n bnc..bnc+3)

    auto load_tile = [&](int k0) {
#pragma unroll
        for (int p = 0; p < 4; ++p)
            ra[p] = *(const float4*)&A[(size_t)(blockM + arow + p * 32) * (size_t)K
                                       + (size_t)(k0 + akc)];
#pragma unroll
        for (int j = 0; j < 4; ++j)
            rb[j] = *(const float4*)&Bmat[(size_t)(k0 + bkc + j) * (size_t)N
                                          + (size_t)(blockN + bnc)];
    };

    auto store_tile = [&](int buf, int k0) {
        float4 pb;
        if (SUBK) pb = *(const float4*)&kbias[k0 + akc];
#pragma unroll
        for (int p = 0; p < 4; ++p) {
            float4 f = ra[p];
            if (SUBK) { f.x -= pb.x; f.y -= pb.y; f.z -= pb.z; f.w -= pb.w; }
            BH4 h{(__bf16)f.x, (__bf16)f.y, (__bf16)f.z, (__bf16)f.w};
            *(BH4*)&As[buf][arow + p * 32][akc] = h;
        }
        // pack 4 consecutive K values per 8-byte store: Bs[n][bkc..bkc+3]
#pragma unroll
        for (int c = 0; c < 4; ++c) {
            BH4 h{(__bf16)f4c(rb[0], c), (__bf16)f4c(rb[1], c),
                  (__bf16)f4c(rb[2], c), (__bf16)f4c(rb[3], c)};
            *(BH4*)&Bs[buf][bnc + c][bkc] = h;
        }
    };

    // prologue: stage tile 0
    load_tile(0);
    store_tile(0, 0);

    const int niter = K / BK;
    for (int it = 0; it < niter; ++it) {
        __syncthreads();                       // staged buf visible; prev reads done
        const int buf  = it & 1;
        const bool more = (it + 1 < niter);

        if (more) load_tile((it + 1) * BK);    // overlap global latency with WMMAs
        if (it + 2 < niter) {
            __builtin_prefetch(&A[(size_t)(blockM + arow) * (size_t)K
                                  + (size_t)((it + 2) * BK + akc)], 0, 0);
            __builtin_prefetch(&Bmat[(size_t)((it + 2) * BK + bkc) * (size_t)N
                                     + (size_t)(blockN + bnc)], 0, 0);
        }

        // ---- fragment loads (16-bit A 16x32 / B 32x16 WMMA VGPR layouts)
        FragU af[4];
#pragma unroll
        for (int i = 0; i < 4; ++i) {
            const int r = waveM * 64 + i * 16 + l16;
            af[i].q[0] = *(const uint4*)&As[buf][r][8 * half];       // K 0..7 | 8..15
            af[i].q[1] = *(const uint4*)&As[buf][r][16 + 8 * half];  // K 16..23 | 24..31
        }
        FragU bfr[2];
#pragma unroll
        for (int j = 0; j < 2; ++j) {
            const int c = waveN * 32 + j * 16 + l16;
            bfr[j].q[0] = *(const uint4*)&Bs[buf][c][8 * half];      // K 0..7 | 8..15
            bfr[j].q[1] = *(const uint4*)&Bs[buf][c][16 + 8 * half]; // K 16..23 | 24..31
        }

#pragma unroll
        for (int i = 0; i < 4; ++i)
#pragma unroll
            for (int j = 0; j < 2; ++j)
                acc[i][j] = __builtin_amdgcn_wmma_f32_16x16x32_bf16(
                    /*neg_a=*/false, af[i].v, /*neg_b=*/false, bfr[j].v,
                    /*c_mod=*/(short)0, acc[i][j],
                    /*reuse_a=*/false, /*reuse_b=*/false);

        if (more) store_tile(buf ^ 1, (it + 1) * BK);
    }

    // ---- store: C/D layout -> VGPR r holds (M=r | M=r+8) x (N=lane&15)
#pragma unroll
    for (int i = 0; i < 4; ++i) {
        const int row0 = blockM + waveM * 64 + i * 16 + half * 8;
#pragma unroll
        for (int j = 0; j < 2; ++j) {
            const int col = blockN + waveN * 32 + j * 16 + l16;
            const float cb = cbias[col];
#pragma unroll
            for (int r = 0; r < 8; ++r) {
                float v = acc[i][j][r] + cb;
                if (RELU) v = fmaxf(v, 0.f);
                C[(size_t)(row0 + r) * (size_t)N + (size_t)col] = v;
            }
        }
    }
}

// ---------------------------------------------------------------------------
// Top-32 per row over full_latents (already relu'd, so >= 0; monotone relu =>
// top-32 of relu(pre) yields identical sparse latents as relu(top-32(pre))).
// One block per row; row cached in dynamic LDS; 32 deterministic argmax rounds.
// ---------------------------------------------------------------------------
__global__ __launch_bounds__(256)
void topk32_kernel(const float* __restrict__ fl,   // [B][L]
                   float* __restrict__ latents,    // [B][L] (zeroed + scattered)
                   float* __restrict__ tv,         // [B][32]
                   int* __restrict__ ti) {         // [B][32]
    const int L = 16384;
    const int row = blockIdx.x;
    const int tid = threadIdx.x;   // 256

    extern __shared__ float srow[];          // L floats (64 KB)
    __shared__ float redv[256];
    __shared__ int   redi[256];
    __shared__ float selv[32];
    __shared__ int   seli[32];

    const float* src = fl + (size_t)row * L;
    float*       dst = latents + (size_t)row * L;

    for (int i = tid * 4; i < L; i += 256 * 4) {
        const float4 v = *(const float4*)&src[i];
        *(float4*)&srow[i] = v;
        *(float4*)&dst[i]  = make_float4(0.f, 0.f, 0.f, 0.f);
    }
    __syncthreads();

    for (int s = 0; s < 32; ++s) {
        float bv = -1.f;
        int   bi = 0;
        for (int i = tid; i < L; i += 256) {
            const float v = srow[i];
            if (v > bv) { bv = v; bi = i; }
        }
        redv[tid] = bv; redi[tid] = bi;
        __syncthreads();
#pragma unroll
        for (int off = 128; off > 0; off >>= 1) {
            if (tid < off) {
                const float ov = redv[tid + off];
                const int   oi = redi[tid + off];
                if (ov > redv[tid] || (ov == redv[tid] && oi < redi[tid])) {
                    redv[tid] = ov; redi[tid] = oi;
                }
            }
            __syncthreads();
        }
        if (tid == 0) {
            selv[s] = redv[0];
            seli[s] = redi[0];
            srow[redi[0]] = -1.f;   // mask (all live values >= 0)
        }
        __syncthreads();
    }

    if (tid < 32) {
        const float v = selv[tid];
        const int   i = seli[tid];
        dst[i] = v;
        tv[(size_t)row * 32 + tid] = v;
        ti[(size_t)row * 32 + tid] = i;
    }
}

// ---------------------------------------------------------------------------
// reconstructed[row] = pre_bias + sum_j val_j * decoder[idx_j, :]
// decoder (134 MB) is L2-resident; per-block traffic 32 rows x 8 KB.
// ---------------------------------------------------------------------------
__global__ __launch_bounds__(256)
void sparse_decode_kernel(const float* __restrict__ tv,
                          const int* __restrict__ ti,
                          const float* __restrict__ dec,      // [L][D]
                          const float* __restrict__ pre_bias, // [D]
                          float* __restrict__ rec) {          // [B][D]
    const int D = 2048;
    const int row = blockIdx.x;
    const int tid = threadIdx.x;   // 256 -> 8 cols each

    __shared__ float sv[32];
    __shared__ int   si[32];
    if (tid < 32) {
        sv[tid] = tv[(size_t)row * 32 + tid];
        si[tid] = ti[(size_t)row * 32 + tid];
    }
    __syncthreads();

    const int c0 = tid * 8;
    float4 a0 = *(const float4*)&pre_bias[c0];
    float4 a1 = *(const float4*)&pre_bias[c0 + 4];

#pragma unroll 4
    for (int j = 0; j < 32; ++j) {
        const float  w    = sv[j];
        const float* drow = dec + (size_t)si[j] * D + c0;
        const float4 d0 = *(const float4*)drow;
        const float4 d1 = *(const float4*)(drow + 4);
        a0.x = fmaf(w, d0.x, a0.x); a0.y = fmaf(w, d0.y, a0.y);
        a0.z = fmaf(w, d0.z, a0.z); a0.w = fmaf(w, d0.w, a0.w);
        a1.x = fmaf(w, d1.x, a1.x); a1.y = fmaf(w, d1.y, a1.y);
        a1.z = fmaf(w, d1.z, a1.z); a1.w = fmaf(w, d1.w, a1.w);
    }

    float* out = rec + (size_t)row * D + c0;
    *(float4*)out       = a0;
    *(float4*)(out + 4) = a1;
}

// ---------------------------------------------------------------------------
extern "C" void kernel_launch(void* const* d_in, const int* in_sizes, int n_in,
                              void* d_out, int out_size, void* d_ws, size_t ws_size,
                              hipStream_t stream) {
    (void)in_sizes; (void)n_in; (void)out_size; (void)ws_size;

    const float* x        = (const float*)d_in[0];  // [8192,2048]
    const float* pre_bias = (const float*)d_in[1];  // [2048]
    const float* enc      = (const float*)d_in[2];  // [2048,16384]
    const float* lat_bias = (const float*)d_in[3];  // [16384]
    const float* dec      = (const float*)d_in[4];  // [16384,2048]
    // d_in[5] = k (fixed = 32 for this problem size)

    const int B = 8192, D = 2048, L = 16384;

    float* out       = (float*)d_out;
    float* rec       = out;                                   // [B,D]
    float* latents   = rec + (size_t)B * D;                   // [B,L]
    float* rec_full  = latents + (size_t)B * L;               // [B,D]
    float* full_lat  = rec_full + (size_t)B * D;              // [B,L]

    float* tv = (float*)d_ws;                                 // [B,32]
    int*   ti = (int*)((char*)d_ws + (size_t)B * 32 * sizeof(float));

    const dim3 blk(256);

    // 1) full_latents = relu((x - pre_bias) @ enc + lat_bias)
    gemm_bf16_wmma<true, true>
        <<<dim3(L / BN, B / BM), blk, 0, stream>>>(x, enc, full_lat,
                                                   pre_bias, lat_bias, B, L, D);

    // 2) top-32 per row -> sparse latents + compact (val, idx)
    topk32_kernel<<<dim3(B), blk, 16384 * sizeof(float), stream>>>(
        full_lat, latents, tv, ti);

    // 3) reconstructed_full = full_latents @ dec + pre_bias
    gemm_bf16_wmma<false, false>
        <<<dim3(D / BN, B / BM), blk, 0, stream>>>(full_lat, dec, rec_full,
                                                   nullptr, pre_bias, B, D, L);

    // 4) reconstructed = latents @ dec + pre_bias  (sparse gather)
    sparse_decode_kernel<<<dim3(B), blk, 0, stream>>>(tv, ti, dec, pre_bias, rec);
}
